// NTM_50689204028032
// MI455X (gfx1250) — compile-verified, matrix operand order checked by reading, not run on hardware
//
#include <hip/hip_runtime.h>
#include <math.h>

typedef __attribute__((ext_vector_type(16))) _Float16 v16h;
typedef __attribute__((ext_vector_type(8)))  float    v8f;

#define B_    128
#define N_    4096
#define M_    128
#define H_    512
#define OUT_  128
#define G_    2048   // 4*H
#define KX_   768    // INPUT + M + H
#define EPS_  1e-8f

__device__ __forceinline__ float sigf(float x)      { return 1.0f / (1.0f + __expf(-x)); }
__device__ __forceinline__ float softplusf(float x) { return (x > 20.0f) ? x : log1pf(__expf(x)); }

// -------------------------------------------------------------------------
// WMMA fragment K-striping for 16-bit A/B (wave32):
// lane<16 -> K-half base 0, lane>=16 -> base 8; pairs per VGPR:
// i<4: K = base+2i ; i>=4: K = base+16+2(i-4)  (and +1 in high half)
__device__ __forceinline__ int koff_of(int i, int half) {
    return half * 8 + ((i < 4) ? 2 * i : 16 + 2 * (i - 4));
}

// -------------------------------------------------------------------------
// Kernel A: gates = [inp | prev_reads | h_prev] (128x768) @ [W_ih|W_hh]^T (768x2048)
// one 16(m)x128(n) super-tile per block; 8 waves, one 16x16 tile each.
__global__ void gates_gemm(const float* __restrict__ inp,
                           const float* __restrict__ prev_reads,
                           const float* __restrict__ h_prev,
                           const float* __restrict__ W_ih,
                           const float* __restrict__ W_hh,
                           float* __restrict__ gates) {
    __shared__ float As[16 * 32];
    __shared__ float Bs[128 * 32];   // [j_local][k]
    const int tid  = threadIdx.x;
    const int wave = tid >> 5;
    const int lane = tid & 31;
    const int half = lane >> 4;
    const int mn   = lane & 15;
    const int mBase = blockIdx.y * 16;
    const int nBase = blockIdx.x * 128;
    const int nCol  = nBase + wave * 16 + mn;

    v8f acc = {0.f,0.f,0.f,0.f,0.f,0.f,0.f,0.f};
    for (int k0 = 0; k0 < KX_; k0 += 32) {
        { // stage A chunk 16x32 (concat of 3 sources)
            int idx = tid * 2;
            int ml = idx >> 5, kk = idx & 31;
            int m = mBase + ml;
#pragma unroll
            for (int q = 0; q < 2; ++q) {
                int k = k0 + kk + q;
                float v;
                if      (k < 128) v = inp[m * 128 + k];
                else if (k < 256) v = prev_reads[m * 128 + (k - 128)];
                else              v = h_prev[m * 512 + (k - 256)];
                As[ml * 32 + kk + q] = v;
            }
        }
        { // stage B chunk: Bs[j_local][k] = W[j, k]  (W rows contiguous in k)
            int jl = tid >> 1;
            int kk0 = (tid & 1) * 16;
            int j = nBase + jl;
#pragma unroll
            for (int q = 0; q < 16; ++q) {
                int k = k0 + kk0 + q;
                float v = (k < 256) ? W_ih[j * 256 + k] : W_hh[j * 512 + (k - 256)];
                Bs[jl * 32 + kk0 + q] = v;
            }
        }
        __syncthreads();
        v16h af, bf;
        int jl = wave * 16 + mn;
#pragma unroll
        for (int i = 0; i < 8; ++i) {
            int ko = koff_of(i, half);
            af[2*i]   = (_Float16)As[mn * 32 + ko];
            af[2*i+1] = (_Float16)As[mn * 32 + ko + 1];
            bf[2*i]   = (_Float16)Bs[jl * 32 + ko];
            bf[2*i+1] = (_Float16)Bs[jl * 32 + ko + 1];
        }
        acc = __builtin_amdgcn_wmma_f32_16x16x32_f16(false, af, false, bf,
                                                     (short)0, acc, false, false);
        __syncthreads();
    }
#pragma unroll
    for (int r = 0; r < 8; ++r) {
        int m = mBase + half * 8 + r;
        gates[m * G_ + nCol] = acc[r];
    }
}

// -------------------------------------------------------------------------
// Kernel C: Out(128 x Ntot) = h(128x512) @ W(512 x Ntot, row-major), edge-guarded N.
__global__ void small_gemm(const float* __restrict__ A,
                           const float* __restrict__ W,
                           float* __restrict__ Out, int Ntot) {
    __shared__ float As[16 * 32];
    __shared__ float Bs[32 * 128];   // [k][n_local]
    const int tid  = threadIdx.x;
    const int wave = tid >> 5;
    const int lane = tid & 31;
    const int half = lane >> 4;
    const int mn   = lane & 15;
    const int mBase = blockIdx.y * 16;
    const int nBase = blockIdx.x * 128;
    const int nCol  = nBase + wave * 16 + mn;

    v8f acc = {0.f,0.f,0.f,0.f,0.f,0.f,0.f,0.f};
    for (int k0 = 0; k0 < H_; k0 += 32) {
        {
            int idx = tid * 2;
            int ml = idx >> 5, kk = idx & 31;
            As[ml * 32 + kk]     = A[(mBase + ml) * H_ + k0 + kk];
            As[ml * 32 + kk + 1] = A[(mBase + ml) * H_ + k0 + kk + 1];
        }
        {
            int kl = tid >> 3;
            int noff = (tid & 7) * 16;
#pragma unroll
            for (int q = 0; q < 16; ++q) {
                int n = nBase + noff + q;
                Bs[kl * 128 + noff + q] = (n < Ntot) ? W[(k0 + kl) * Ntot + n] : 0.0f;
            }
        }
        __syncthreads();
        v16h af, bf;
        int nl = wave * 16 + mn;
#pragma unroll
        for (int i = 0; i < 8; ++i) {
            int ko = koff_of(i, half);
            af[2*i]   = (_Float16)As[mn * 32 + ko];
            af[2*i+1] = (_Float16)As[mn * 32 + ko + 1];
            bf[2*i]   = (_Float16)Bs[ko * 128 + nl];
            bf[2*i+1] = (_Float16)Bs[(ko + 1) * 128 + nl];
        }
        acc = __builtin_amdgcn_wmma_f32_16x16x32_f16(false, af, false, bf,
                                                     (short)0, acc, false, false);
        __syncthreads();
    }
    if (nCol < Ntot) {
#pragma unroll
        for (int r = 0; r < 8; ++r) {
            int m = mBase + half * 8 + r;
            Out[m * Ntot + nCol] = acc[r];
        }
    }
}

// -------------------------------------------------------------------------
__global__ void lstm_pointwise(const float* __restrict__ gates,
                               const float* __restrict__ b_ih,
                               const float* __restrict__ b_hh,
                               const float* __restrict__ c_prev,
                               float* __restrict__ h_out,
                               float* __restrict__ c_out) {
    int idx = blockIdx.x * blockDim.x + threadIdx.x;
    if (idx >= B_ * H_) return;
    int b = idx / H_, j = idx - b * H_;
    const float* gr = gates + b * G_;
    float ig = gr[j]          + b_ih[j]          + b_hh[j];
    float fg = gr[H_ + j]     + b_ih[H_ + j]     + b_hh[H_ + j];
    float gg = gr[2*H_ + j]   + b_ih[2*H_ + j]   + b_hh[2*H_ + j];
    float og = gr[3*H_ + j]   + b_ih[3*H_ + j]   + b_hh[3*H_ + j];
    float c = sigf(fg) * c_prev[idx] + sigf(ig) * tanhf(gg);
    float h = sigf(og) * tanhf(c);
    h_out[idx] = h;
    c_out[idx] = c;
}

// -------------------------------------------------------------------------
// per-batch head params: k = tanh(p[:M]), ||k||, beta/g/s/gamma
__global__ void split_addr(const float* __restrict__ P, const float* __restrict__ bias,
                           int Pstride, float* __restrict__ kvec,
                           float* __restrict__ params) {
    __shared__ float red[128];
    int b = blockIdx.x, t = threadIdx.x;
    const float* p = P + b * Pstride;
    float kv = tanhf(p[t] + bias[t]);
    kvec[b * M_ + t] = kv;
    red[t] = kv * kv;
    __syncthreads();
    for (int s = 64; s > 0; s >>= 1) { if (t < s) red[t] += red[t + s]; __syncthreads(); }
    if (t == 0) {
        float kn = sqrtf(red[0]);
        float beta = softplusf(p[M_]     + bias[M_]);
        float g    = sigf(p[M_ + 1] + bias[M_ + 1]);
        float a0 = p[M_+2]+bias[M_+2], a1 = p[M_+3]+bias[M_+3], a2 = p[M_+4]+bias[M_+4];
        float mx = fmaxf(a0, fmaxf(a1, a2));
        float e0 = __expf(a0-mx), e1 = __expf(a1-mx), e2 = __expf(a2-mx);
        float es = e0 + e1 + e2;
        float gamma = 1.0f + softplusf(p[M_+5] + bias[M_+5]);
        float* pp = params + b * 8;
        pp[0]=beta; pp[1]=g; pp[2]=e0/es; pp[3]=e1/es; pp[4]=e2/es; pp[5]=gamma; pp[6]=kn;
    }
}

// -------------------------------------------------------------------------
// Pass 1 over memory: one wave per (b,n) row; fused cosine sims for both heads.
__global__ void sim_pass(const float* __restrict__ mem,
                         const float* __restrict__ k_r, const float* __restrict__ k_w,
                         const float* __restrict__ par_r, const float* __restrict__ par_w,
                         float* __restrict__ sim_r, float* __restrict__ sim_w) {
    int wave = (blockIdx.x * blockDim.x + threadIdx.x) >> 5;
    int lane = threadIdx.x & 31;
    int b = wave >> 12;          // /4096
    int n = wave & (N_ - 1);
    const float4* row = (const float4*)(mem + ((size_t)(b * N_ + n)) * M_);
    const float4* kr4 = (const float4*)(k_r + b * M_);
    const float4* kw4 = (const float4*)(k_w + b * M_);
    float4 f = row[lane];
    float4 a = kr4[lane];
    float4 c = kw4[lane];
    float dr = f.x*a.x + f.y*a.y + f.z*a.z + f.w*a.w;
    float dw = f.x*c.x + f.y*c.y + f.z*c.z + f.w*c.w;
    float nn = f.x*f.x + f.y*f.y + f.z*f.z + f.w*f.w;
    for (int o = 16; o > 0; o >>= 1) {
        dr += __shfl_xor(dr, o, 32);
        dw += __shfl_xor(dw, o, 32);
        nn += __shfl_xor(nn, o, 32);
    }
    if (lane == 0) {
        float rn = sqrtf(nn);
        sim_r[b * N_ + n] = dr / (rn * par_r[b * 8 + 6] + EPS_);
        sim_w[b * N_ + n] = dw / (rn * par_w[b * 8 + 6] + EPS_);
    }
}

// -------------------------------------------------------------------------
// softmax + interpolate + circular shift + sharpen + normalize; one block per (b,head)
__global__ void address_finish(const float* __restrict__ sim_r, const float* __restrict__ sim_w,
                               const float* __restrict__ par_r, const float* __restrict__ par_w,
                               const float* __restrict__ prev_r, const float* __restrict__ prev_w,
                               float* __restrict__ out_r, float* __restrict__ out_w) {
    __shared__ float wg[N_];
    __shared__ float red[1024];
    int b = blockIdx.x, rw = blockIdx.y, t = threadIdx.x;
    const float* sim  = rw ? sim_w : sim_r;
    const float* par  = (rw ? par_w : par_r) + b * 8;
    const float* prev = rw ? prev_w : prev_r;
    float* out        = rw ? out_w : out_r;
    float beta = par[0], g = par[1], s0 = par[2], s1 = par[3], s2 = par[4], gamma = par[5];

    float v[4], lmax = -1e30f;
#pragma unroll
    for (int i = 0; i < 4; ++i) { v[i] = beta * sim[b * N_ + t * 4 + i]; lmax = fmaxf(lmax, v[i]); }
    red[t] = lmax; __syncthreads();
    for (int s = 512; s > 0; s >>= 1) { if (t < s) red[t] = fmaxf(red[t], red[t + s]); __syncthreads(); }
    float mx = red[0]; __syncthreads();

    float lsum = 0.f;
#pragma unroll
    for (int i = 0; i < 4; ++i) { v[i] = __expf(v[i] - mx); lsum += v[i]; }
    red[t] = lsum; __syncthreads();
    for (int s = 512; s > 0; s >>= 1) { if (t < s) red[t] += red[t + s]; __syncthreads(); }
    float inv = 1.0f / red[0]; __syncthreads();

#pragma unroll
    for (int i = 0; i < 4; ++i) {
        int n = t * 4 + i;
        wg[n] = g * v[i] * inv + (1.0f - g) * prev[b * N_ + n];
    }
    __syncthreads();

    float lsum2 = 0.f;
#pragma unroll
    for (int i = 0; i < 4; ++i) {
        int n = t * 4 + i;
        float wt = s0 * wg[(n + N_ - 1) & (N_ - 1)] + s1 * wg[n] + s2 * wg[(n + 1) & (N_ - 1)];
        v[i] = __expf(gamma * __logf(wt + EPS_));
        lsum2 += v[i];
    }
    red[t] = lsum2; __syncthreads();
    for (int s = 512; s > 0; s >>= 1) { if (t < s) red[t] += red[t + s]; __syncthreads(); }
    float inv2 = 1.0f / red[0];
#pragma unroll
    for (int i = 0; i < 4; ++i) out[b * N_ + t * 4 + i] = v[i] * inv2;
}

// -------------------------------------------------------------------------
// Pass 2: new_memory + read_vec accumulation. Block = (b, 128-row chunk); 8 waves x 16 rows.
__global__ void memory_update(const float* __restrict__ mem,
                              const float* __restrict__ wp, const float* __restrict__ write_b,
                              const float* __restrict__ read_w, const float* __restrict__ write_w,
                              float* __restrict__ new_mem, float* __restrict__ read_vec) {
    __shared__ float red[8 * 128];
    int b = blockIdx.y, chunk = blockIdx.x;
    int wave = threadIdx.x >> 5, lane = threadIdx.x & 31;
    int m0 = lane * 4;
    const float* wpb = wp + b * 390;
    float4 e4, a4;
    e4.x = sigf(wpb[134 + m0]     + write_b[134 + m0]);
    e4.y = sigf(wpb[134 + m0 + 1] + write_b[134 + m0 + 1]);
    e4.z = sigf(wpb[134 + m0 + 2] + write_b[134 + m0 + 2]);
    e4.w = sigf(wpb[134 + m0 + 3] + write_b[134 + m0 + 3]);
    a4.x = tanhf(wpb[262 + m0]     + write_b[262 + m0]);
    a4.y = tanhf(wpb[262 + m0 + 1] + write_b[262 + m0 + 1]);
    a4.z = tanhf(wpb[262 + m0 + 2] + write_b[262 + m0 + 2]);
    a4.w = tanhf(wpb[262 + m0 + 3] + write_b[262 + m0 + 3]);

    float4 acc = {0.f, 0.f, 0.f, 0.f};
    for (int r = 0; r < 16; ++r) {
        int n = chunk * 128 + wave * 16 + r;
        size_t off = ((size_t)(b * N_ + n)) * M_;
        float ww = write_w[b * N_ + n];
        float wr = read_w[b * N_ + n];
        float4 f = ((const float4*)(mem + off))[lane];
        float4 o;
        o.x = f.x * (1.0f - ww * e4.x) + ww * a4.x;
        o.y = f.y * (1.0f - ww * e4.y) + ww * a4.y;
        o.z = f.z * (1.0f - ww * e4.z) + ww * a4.z;
        o.w = f.w * (1.0f - ww * e4.w) + ww * a4.w;
        ((float4*)(new_mem + off))[lane] = o;
        acc.x += wr * f.x; acc.y += wr * f.y; acc.z += wr * f.z; acc.w += wr * f.w;
    }
    red[wave * 128 + m0]     = acc.x;
    red[wave * 128 + m0 + 1] = acc.y;
    red[wave * 128 + m0 + 2] = acc.z;
    red[wave * 128 + m0 + 3] = acc.w;
    __syncthreads();
    if (threadIdx.x < 128) {
        float s = 0.f;
#pragma unroll
        for (int w2 = 0; w2 < 8; ++w2) s += red[w2 * 128 + threadIdx.x];
        atomicAdd(&read_vec[b * M_ + threadIdx.x], s);
    }
}

// -------------------------------------------------------------------------
__global__ void fc_out(const float* __restrict__ h, const float* __restrict__ rv,
                       const float* __restrict__ fc_W, const float* __restrict__ fc_b,
                       float* __restrict__ out) {
    __shared__ float xh[H_ + M_];
    int b = blockIdx.x, t = threadIdx.x;
    for (int k = t; k < H_; k += 128) xh[k] = h[b * H_ + k];
    xh[H_ + t] = rv[b * M_ + t];
    __syncthreads();
    float acc = fc_b[t];
    for (int k = 0; k < H_ + M_; ++k) acc += xh[k] * fc_W[k * OUT_ + t];
    out[b * OUT_ + t] = sigf(acc);
}

__global__ void zero_kernel(float* __restrict__ p, int n) {
    int i = blockIdx.x * blockDim.x + threadIdx.x;
    if (i < n) p[i] = 0.0f;
}

// -------------------------------------------------------------------------
extern "C" void kernel_launch(void* const* d_in, const int* in_sizes, int n_in,
                              void* d_out, int out_size, void* d_ws, size_t ws_size,
                              hipStream_t stream) {
    const float* inp        = (const float*)d_in[0];
    const float* prev_reads = (const float*)d_in[1];
    const float* h_prev     = (const float*)d_in[2];
    const float* c_prev     = (const float*)d_in[3];
    const float* prev_rw    = (const float*)d_in[4];
    const float* prev_ww    = (const float*)d_in[5];
    const float* memory     = (const float*)d_in[6];
    const float* W_ih       = (const float*)d_in[7];
    const float* W_hh       = (const float*)d_in[8];
    const float* b_ih       = (const float*)d_in[9];
    const float* b_hh       = (const float*)d_in[10];
    const float* read_W     = (const float*)d_in[11];
    const float* read_b     = (const float*)d_in[12];
    const float* write_W    = (const float*)d_in[13];
    const float* write_b    = (const float*)d_in[14];
    const float* fc_W       = (const float*)d_in[15];
    const float* fc_b       = (const float*)d_in[16];

    float* out = (float*)d_out;
    float* o_out = out;                  // B*OUT        = 16384
    float* o_rv  = out + 16384;          // B*M          = 16384
    float* o_h   = out + 32768;          // B*H          = 65536
    float* o_c   = out + 98304;          // B*H          = 65536
    float* o_rw  = out + 163840;         // B*N          = 524288
    float* o_ww  = out + 688128;         // B*N          = 524288
    float* o_nm  = out + 1212416;        // B*N*M

    float* ws    = (float*)d_ws;
    float* gates = ws;                     // 128*2048 = 262144
    float* rp    = gates + 262144;         // 128*134  = 17152
    float* wp    = rp + 17152;             // 128*390  = 49920
    float* k_r   = wp + 49920;             // 16384
    float* k_w   = k_r + 16384;            // 16384
    float* par_r = k_w + 16384;            // 1024
    float* par_w = par_r + 1024;           // 1024
    float* sim_r = par_w + 1024;           // 524288
    float* sim_w = sim_r + 524288;         // 524288   (total ~5.7 MB)

    zero_kernel<<<64, 256, 0, stream>>>(o_rv, B_ * M_);

    gates_gemm<<<dim3(16, 8), 256, 0, stream>>>(inp, prev_reads, h_prev, W_ih, W_hh, gates);
    lstm_pointwise<<<256, 256, 0, stream>>>(gates, b_ih, b_hh, c_prev, o_h, o_c);

    small_gemm<<<dim3(2, 8), 256, 0, stream>>>(o_h, read_W,  rp, 134);
    small_gemm<<<dim3(4, 8), 256, 0, stream>>>(o_h, write_W, wp, 390);

    split_addr<<<128, 128, 0, stream>>>(rp, read_b,  134, k_r, par_r);
    split_addr<<<128, 128, 0, stream>>>(wp, write_b, 390, k_w, par_w);

    sim_pass<<<65536, 256, 0, stream>>>(memory, k_r, k_w, par_r, par_w, sim_r, sim_w);

    address_finish<<<dim3(128, 2), 1024, 0, stream>>>(sim_r, sim_w, par_r, par_w,
                                                      prev_rw, prev_ww, o_rw, o_ww);

    memory_update<<<dim3(32, 128), 256, 0, stream>>>(memory, wp, write_b, o_rw, o_ww,
                                                     o_nm, o_rv);

    fc_out<<<128, 128, 0, stream>>>(o_h, o_rv, fc_W, fc_b, o_out);
}
